// decoder_with_attention_net_36885179138473
// MI455X (gfx1250) — matrix operand order repaired and checked by reading, not compiled, not run on hardware
//
#include <hip/hip_runtime.h>
#include <hip/hip_bf16.h>
#include <cstdint>
#include <cstddef>

// ---------------- problem dimensions ----------------
constexpr int NB = 64;          // batch
constexpr int NT = 52;          // caption length
constexpr int NP = 196;         // pixels
constexpr int NV = 10000;       // vocab
constexpr int NA = 512;         // attention dim
constexpr int NE = 512;         // embed dim
constexpr int NH = 512;         // hidden
constexpr int ND = 2048;        // encoder dim
constexpr int NSTEPS = NT - 1;  // 51 decode steps
constexpr int NX = NE + ND;     // 2560 LSTM input
constexpr int NG = 4 * NH;      // 2048 gates

// ---------------- types for WMMA (gfx1250, wave32) ----------------
typedef __attribute__((ext_vector_type(16))) __bf16       v16bf;
typedef __attribute__((ext_vector_type(8)))  float        v8f;
typedef __attribute__((ext_vector_type(4)))  unsigned int u32x4;

union FragU { u32x4 q[2]; v16bf bf; };

__device__ __forceinline__ unsigned short f2bf(float f) {
    unsigned int u = __float_as_uint(f);
    u += 0x7FFFu + ((u >> 16) & 1u);   // round-to-nearest-even
    return (unsigned short)(u >> 16);
}
__device__ __forceinline__ float bf2f(unsigned short s) {
    return __uint_as_float(((unsigned int)s) << 16);
}
__device__ __forceinline__ float sigmoidf_(float x) {
    return 1.0f / (1.0f + __expf(-x));
}

// ---------------- generic bf16 WMMA GEMM, 4 M-tiles per wave ----------------
// C(M,N) = A(M,K)[bf16] @ W(N,K)^T[bf16] + bias(N) [+ Cin] , then * rowscale[m]
// Each wave owns one 16-wide N-tile and MT(=4) stacked 16-row M-tiles:
// the weight (B) fragment is loaded ONCE per K-step and reused for MT WMMAs,
// quartering weight traffic versus one-tile-per-wave. 8 waves/block along N.
// A-fragment per documented 16-bit A layout (lane L: m=L%16, g=L/16;
// elems 0-7 = K[kt+8g..+8), elems 8-15 = K[kt+16+8g..+8)).
// B-fragment: lane L: n=L%16, g=L/16; elems 0-7 = K[kt+8g..+8), elems 8-15 = K[kt+16+8g..+8) of W row n.
template <int MT>
__global__ __launch_bounds__(256)
void gemm_bf16_wmma_mt(const unsigned short* __restrict__ A, int lda,
                       const unsigned short* __restrict__ Bw, int ldb,
                       const float* __restrict__ bias,
                       const float* __restrict__ Cin, int ldcin,
                       const float* __restrict__ rowscale,
                       float* __restrict__ C, int ldc,
                       int M, int N, int K)
{
    const int lane = threadIdx.x & 31;
    const int wave = threadIdx.x >> 5;
    const int nt   = blockIdx.x * 8 + wave;
    const int n0   = nt * 16;
    if (n0 >= N) return;
    const int m0 = blockIdx.y * 16 * MT;
    const int g  = lane >> 4;
    const int l  = lane & 15;

    const unsigned short* arow = A  + (size_t)(m0 + l) * lda;
    const unsigned short* brow = Bw + (size_t)(n0 + l) * ldb;

    v8f acc[MT];
#pragma unroll
    for (int mt = 0; mt < MT; ++mt) acc[mt] = (v8f){0.f,0.f,0.f,0.f,0.f,0.f,0.f,0.f};

    for (int kt = 0; kt < K; kt += 32) {
        FragU fb;
        fb.q[0] = *reinterpret_cast<const u32x4*>(brow + kt + g * 8);
        fb.q[1] = *reinterpret_cast<const u32x4*>(brow + kt + 16 + g * 8);
#pragma unroll
        for (int mt = 0; mt < MT; ++mt) {
            const unsigned short* am = arow + (size_t)mt * 16 * lda;
            FragU fa;
            fa.q[0] = *reinterpret_cast<const u32x4*>(am + kt + g * 8);
            fa.q[1] = *reinterpret_cast<const u32x4*>(am + kt + 16 + g * 8);
            acc[mt] = __builtin_amdgcn_wmma_f32_16x16x32_bf16(
                false, fa.bf, false, fb.bf, (short)0, acc[mt], false, false);
        }
    }

    const int n = n0 + l;
    const float badd = bias ? bias[n] : 0.0f;
#pragma unroll
    for (int mt = 0; mt < MT; ++mt) {
#pragma unroll
        for (int r = 0; r < 8; ++r) {
            const int m = m0 + mt * 16 + g * 8 + r;   // C layout: VGPR r -> M = r + 8*(lane/16)
            if (m < M) {
                float v = acc[mt][r] + badd;
                if (Cin)      v += Cin[(size_t)m * ldcin + n];
                if (rowscale) v *= rowscale[m];
                C[(size_t)m * ldc + n] = v;
            }
        }
    }
}

// ---------------- preamble kernels ----------------
__global__ __launch_bounds__(64)
void order_kernel(const int* __restrict__ caps_len,
                  int* __restrict__ order_i, int* __restrict__ lens_s,
                  float* __restrict__ out_lens, float* __restrict__ out_order)
{
    const int i = threadIdx.x;
    if (i >= NB) return;
    const int li = caps_len[i];
    int rank = 0;
    for (int j = 0; j < NB; ++j) {
        const int lj = caps_len[j];
        if (lj > li || (lj == li && j < i)) rank++;     // stable descending
    }
    order_i[rank] = i;
    __syncthreads();
    const int src = order_i[i];
    lens_s[i]    = caps_len[src];
    out_order[i] = (float)src;
    out_lens[i]  = (float)caps_len[src];
}

__global__ __launch_bounds__(256)
void argmax_kernel(const float* __restrict__ caps, const int* __restrict__ order_i,
                   int* __restrict__ capids, float* __restrict__ out_ids)
{
    const int bt = blockIdx.x;          // b*NT + t
    const int bs = bt / NT, t = bt % NT;
    const int src = order_i[bs];
    const float* row = caps + ((size_t)src * NT + t) * NV;
    float best = -3.0e38f; int bidx = NV;
    for (int v = threadIdx.x; v < NV; v += 256) {
        const float val = row[v];
        if (val > best) { best = val; bidx = v; }
    }
    __shared__ float sv[256]; __shared__ int si[256];
    sv[threadIdx.x] = best; si[threadIdx.x] = bidx;
    __syncthreads();
    for (int s = 128; s > 0; s >>= 1) {
        if (threadIdx.x < s) {
            if (sv[threadIdx.x + s] > sv[threadIdx.x] ||
                (sv[threadIdx.x + s] == sv[threadIdx.x] && si[threadIdx.x + s] < si[threadIdx.x])) {
                sv[threadIdx.x] = sv[threadIdx.x + s];
                si[threadIdx.x] = si[threadIdx.x + s];
            }
        }
        __syncthreads();
    }
    if (threadIdx.x == 0) { capids[bt] = si[0]; out_ids[bt] = (float)si[0]; }
}

__global__ __launch_bounds__(256)
void cvt_bf16_kernel(const float* __restrict__ src, unsigned short* __restrict__ dst, int n)
{
    const int i = blockIdx.x * 256 + threadIdx.x;
    if (i < n) dst[i] = f2bf(src[i]);
}

__global__ __launch_bounds__(256)
void permute_cvt_enc_kernel(const float* __restrict__ enc, const int* __restrict__ order_i,
                            unsigned short* __restrict__ out)
{
    const size_t i = (size_t)blockIdx.x * 256 + threadIdx.x;
    if (i >= (size_t)NB * NP * ND) return;
    const int b = (int)(i / ((size_t)NP * ND));
    const size_t rem = i % ((size_t)NP * ND);
    out[i] = f2bf(enc[(size_t)order_i[b] * NP * ND + rem]);
}

__global__ __launch_bounds__(256)
void embed_kernel(const float* __restrict__ emb_table, const int* __restrict__ capids,
                  unsigned short* __restrict__ emb_bf)
{
    const int i = blockIdx.x * 256 + threadIdx.x;   // NB*NSTEPS*NE
    if (i >= NB * NSTEPS * NE) return;
    const int e  = i % NE;
    const int bt = i / NE;
    const int b  = bt / NSTEPS, t = bt % NSTEPS;
    const int id = capids[b * NT + t];
    emb_bf[i] = f2bf(emb_table[(size_t)id * NE + e]);
}

__global__ __launch_bounds__(256)
void img_mean_kernel(const unsigned short* __restrict__ enc,
                     float* __restrict__ mean_f, unsigned short* __restrict__ mean_bf)
{
    const int i = blockIdx.x * 256 + threadIdx.x;   // NB*ND
    if (i >= NB * ND) return;
    const int b = i / ND, d = i % ND;
    const unsigned short* p = enc + (size_t)b * NP * ND + d;
    float s = 0.f;
    for (int pp = 0; pp < NP; ++pp) s += bf2f(p[(size_t)pp * ND]);
    s *= (1.0f / NP);
    mean_f[i] = s; mean_bf[i] = f2bf(s);
}

// ---------------- per-step kernels ----------------
// block per batch row: logits -> softmax -> gated context
__global__ __launch_bounds__(256)
void attention_kernel(const float* __restrict__ img_proj,   // (B,P,A) incl b_img
                      const float* __restrict__ hW,         // (B,A)   incl b_dec
                      const float* __restrict__ W_att,      // (A)
                      const float* __restrict__ b_att,      // (1)
                      const float* __restrict__ gatepre,    // (B,D)   incl b_gate
                      const unsigned short* __restrict__ enc, // (B,P,D) bf16 sorted
                      const int* __restrict__ lens,
                      float* __restrict__ ctx,              // (B,D)
                      float* __restrict__ alphas_out,       // d_out alphas base
                      int t)
{
    __shared__ float sh_hW[NA];
    __shared__ float sh_Wa[NA];
    __shared__ float sh_lg[NP];
    __shared__ float red[256];
    const int b = blockIdx.x, tid = threadIdx.x;

    for (int a = tid; a < NA; a += 256) { sh_hW[a] = hW[b * NA + a]; sh_Wa[a] = W_att[a]; }
    __syncthreads();

    const float* ip = img_proj + (size_t)b * NP * NA;
    if (tid < NP) {
        const float* row = ip + tid * NA;
        float s = 0.f;
        for (int a = 0; a < NA; ++a) {
            const float v = row[a] + sh_hW[a];
            s += (v > 0.f ? v : 0.f) * sh_Wa[a];
        }
        sh_lg[tid] = s + b_att[0];
    }
    __syncthreads();

    // softmax over P
    float lm = -3.0e38f;
    for (int p = tid; p < NP; p += 256) lm = fmaxf(lm, sh_lg[p]);
    red[tid] = lm; __syncthreads();
    for (int s = 128; s > 0; s >>= 1) { if (tid < s) red[tid] = fmaxf(red[tid], red[tid + s]); __syncthreads(); }
    const float mx = red[0]; __syncthreads();
    float ls = 0.f;
    for (int p = tid; p < NP; p += 256) { const float e = __expf(sh_lg[p] - mx); sh_lg[p] = e; ls += e; }
    red[tid] = ls; __syncthreads();
    for (int s = 128; s > 0; s >>= 1) { if (tid < s) red[tid] += red[tid + s]; __syncthreads(); }
    const float inv = 1.0f / red[0]; __syncthreads();
    for (int p = tid; p < NP; p += 256) sh_lg[p] *= inv;
    __syncthreads();

    const float mf = (t < lens[b] - 1) ? 1.0f : 0.0f;
    for (int p = tid; p < NP; p += 256)
        alphas_out[((size_t)b * NSTEPS + t) * NP + p] = sh_lg[p] * mf;

    const unsigned short* eb = enc + (size_t)b * NP * ND;
    for (int d = tid; d < ND; d += 256) {
        float s = 0.f;
        for (int p = 0; p < NP; ++p) s += sh_lg[p] * bf2f(eb[(size_t)p * ND + d]);
        ctx[b * ND + d] = sigmoidf_(gatepre[b * ND + d]) * s;
    }
}

__global__ __launch_bounds__(256)
void build_x_kernel(const unsigned short* __restrict__ emb_bf, const float* __restrict__ ctx,
                    unsigned short* __restrict__ x_bf, int t)
{
    const int i = blockIdx.x * 256 + threadIdx.x;   // NB*NX
    if (i >= NB * NX) return;
    const int b = i / NX, j = i % NX;
    x_bf[i] = (j < NE) ? emb_bf[((size_t)b * NSTEPS + t) * NE + j]
                       : f2bf(ctx[b * ND + (j - NE)]);
}

__global__ __launch_bounds__(256)
void lstm_update_kernel(const float* __restrict__ gates,
                        float* __restrict__ h, float* __restrict__ c,
                        unsigned short* __restrict__ h_bf,
                        unsigned short* __restrict__ hnew_bf,
                        float* __restrict__ maskf,
                        const int* __restrict__ lens, int t)
{
    const int i = blockIdx.x * 256 + threadIdx.x;   // NB*NH
    if (i >= NB * NH) return;
    const int b = i / NH, j = i % NH;
    const float ig = gates[b * NG + j];
    const float fg = gates[b * NG + NH + j];
    const float gg = gates[b * NG + 2 * NH + j];
    const float og = gates[b * NG + 3 * NH + j];
    const float cn = sigmoidf_(fg) * c[i] + sigmoidf_(ig) * tanhf(gg);
    const float hn = sigmoidf_(og) * tanhf(cn);
    hnew_bf[i] = f2bf(hn);
    const bool m = (t < lens[b] - 1);
    const float hm = m ? hn : h[i];
    const float cm = m ? cn : c[i];
    h[i] = hm; c[i] = cm; h_bf[i] = f2bf(hm);
    if (j == 0) maskf[b] = m ? 1.0f : 0.0f;
}

// ---------------- host launch ----------------
static inline size_t align256(size_t x) { return (x + 255) & ~(size_t)255; }

extern "C" void kernel_launch(void* const* d_in, const int* in_sizes, int n_in,
                              void* d_out, int out_size, void* d_ws, size_t ws_size,
                              hipStream_t stream)
{
    (void)in_sizes; (void)n_in; (void)out_size; (void)ws_size;
    const float* enc_image = (const float*)d_in[0];
    const float* enc_caps  = (const float*)d_in[1];
    const int*   caps_len  = (const int*)  d_in[2];
    const float* W_img  = (const float*)d_in[3];  const float* b_img  = (const float*)d_in[4];
    const float* W_dec  = (const float*)d_in[5];  const float* b_dec  = (const float*)d_in[6];
    const float* W_att  = (const float*)d_in[7];  const float* b_att  = (const float*)d_in[8];
    const float* emb_tb = (const float*)d_in[9];
    const float* W_ih   = (const float*)d_in[10]; const float* W_hh   = (const float*)d_in[11];
    const float* b_ih   = (const float*)d_in[12]; const float* b_hh   = (const float*)d_in[13];
    const float* W_h0   = (const float*)d_in[14]; const float* b_h0   = (const float*)d_in[15];
    const float* W_c0   = (const float*)d_in[16]; const float* b_c0   = (const float*)d_in[17];
    const float* W_gate = (const float*)d_in[18]; const float* b_gate = (const float*)d_in[19];
    const float* W_word = (const float*)d_in[20]; const float* b_word = (const float*)d_in[21];

    float* out        = (float*)d_out;
    float* out_preds  = out;                                      // (B,51,V)
    float* out_alphas = out_preds + (size_t)NB * NSTEPS * NV;     // (B,51,P)
    float* out_ids    = out_alphas + (size_t)NB * NSTEPS * NP;    // (B,T)
    float* out_lens   = out_ids + NB * NT;                        // (B)
    float* out_order  = out_lens + NB;                            // (B)

    // workspace arena
    char* w = (char*)d_ws; size_t off = 0;
    auto alloc = [&](size_t bytes) -> void* { void* p = w + off; off = align256(off + bytes); return p; };
    unsigned short* enc_bf   = (unsigned short*)alloc((size_t)NB * NP * ND * 2);
    float*          img_proj = (float*)         alloc((size_t)NB * NP * NA * 4);
    unsigned short* Wimg_bf  = (unsigned short*)alloc((size_t)NA * ND * 2);
    unsigned short* Wdec_bf  = (unsigned short*)alloc((size_t)NA * NH * 2);
    unsigned short* Wh0_bf   = (unsigned short*)alloc((size_t)NH * ND * 2);
    unsigned short* Wc0_bf   = (unsigned short*)alloc((size_t)NH * ND * 2);
    unsigned short* Wgate_bf = (unsigned short*)alloc((size_t)ND * NH * 2);
    unsigned short* Wih_bf   = (unsigned short*)alloc((size_t)NG * NX * 2);
    unsigned short* Whh_bf   = (unsigned short*)alloc((size_t)NG * NH * 2);
    unsigned short* Wword_bf = (unsigned short*)alloc((size_t)NV * NH * 2);
    unsigned short* emb_bf   = (unsigned short*)alloc((size_t)NB * NSTEPS * NE * 2);
    int*   order_i  = (int*)  alloc(NB * 4);
    int*   lens_i   = (int*)  alloc(NB * 4);
    int*   capids_i = (int*)  alloc(NB * NT * 4);
    float* img_mean = (float*)alloc((size_t)NB * ND * 4);
    unsigned short* imean_bf = (unsigned short*)alloc((size_t)NB * ND * 2);
    float* hbuf     = (float*)alloc((size_t)NB * NH * 4);
    float* cbuf     = (float*)alloc((size_t)NB * NH * 4);
    unsigned short* h_bf    = (unsigned short*)alloc((size_t)NB * NH * 2);
    unsigned short* hnew_bf = (unsigned short*)alloc((size_t)NB * NH * 2);
    float* hW       = (float*)alloc((size_t)NB * NA * 4);
    float* gatepre  = (float*)alloc((size_t)NB * ND * 4);
    float* gates    = (float*)alloc((size_t)NB * NG * 4);
    float* ctx      = (float*)alloc((size_t)NB * ND * 4);
    unsigned short* x_bf = (unsigned short*)alloc((size_t)NB * NX * 2);
    float* maskf    = (float*)alloc(NB * 4);

    // 4 stacked M-tiles per wave -> M must be a multiple of 64 (all ours are).
    auto gemm = [&](const unsigned short* Ap, int lda, const unsigned short* Bp, int ldb,
                    const float* bias, const float* Cin, int ldcin, const float* rs,
                    float* Cp, int ldc, int M, int N, int K) {
        dim3 grid((N + 127) / 128, M / 64);
        gemm_bf16_wmma_mt<4><<<grid, dim3(256), 0, stream>>>(Ap, lda, Bp, ldb, bias,
                                                             Cin, ldcin, rs, Cp, ldc, M, N, K);
    };
    auto cvt = [&](const float* s, unsigned short* d, int n) {
        cvt_bf16_kernel<<<(n + 255) / 256, 256, 0, stream>>>(s, d, n);
    };

    // ---- preamble ----
    order_kernel<<<1, 64, 0, stream>>>(caps_len, order_i, lens_i, out_lens, out_order);
    argmax_kernel<<<NB * NT, 256, 0, stream>>>(enc_caps, order_i, capids_i, out_ids);
    {
        const size_t n = (size_t)NB * NP * ND;
        permute_cvt_enc_kernel<<<(unsigned)((n + 255) / 256), 256, 0, stream>>>(enc_image, order_i, enc_bf);
    }
    cvt(W_img,  Wimg_bf,  NA * ND);
    cvt(W_dec,  Wdec_bf,  NA * NH);
    cvt(W_h0,   Wh0_bf,   NH * ND);
    cvt(W_c0,   Wc0_bf,   NH * ND);
    cvt(W_gate, Wgate_bf, ND * NH);
    cvt(W_ih,   Wih_bf,   NG * NX);
    cvt(W_hh,   Whh_bf,   NG * NH);
    cvt(W_word, Wword_bf, NV * NH);
    embed_kernel<<<(NB * NSTEPS * NE + 255) / 256, 256, 0, stream>>>(emb_tb, capids_i, emb_bf);
    img_mean_kernel<<<(NB * ND + 255) / 256, 256, 0, stream>>>(enc_bf, img_mean, imean_bf);

    // h0 = mean @ W_h0^T + b_h0 ; c0 likewise
    gemm(imean_bf, ND, Wh0_bf, ND, b_h0, nullptr, 0, nullptr, hbuf, NH, NB, NH, ND);
    gemm(imean_bf, ND, Wc0_bf, ND, b_c0, nullptr, 0, nullptr, cbuf, NH, NB, NH, ND);
    cvt(hbuf, h_bf, NB * NH);

    // img_proj = enc_s @ W_img^T + b_img   (M = B*P = 12544)
    gemm(enc_bf, ND, Wimg_bf, ND, b_img, nullptr, 0, nullptr, img_proj, NA, NB * NP, NA, ND);

    // ---- 51 recurrent steps ----
    for (int t = 0; t < NSTEPS; ++t) {
        gemm(h_bf, NH, Wdec_bf,  NH, b_dec,  nullptr, 0, nullptr, hW,      NA, NB, NA, NH);
        gemm(h_bf, NH, Wgate_bf, NH, b_gate, nullptr, 0, nullptr, gatepre, ND, NB, ND, NH);
        gemm(h_bf, NH, Whh_bf,   NH, b_hh,   nullptr, 0, nullptr, gates,   NG, NB, NG, NH);
        attention_kernel<<<NB, 256, 0, stream>>>(img_proj, hW, W_att, b_att, gatepre,
                                                 enc_bf, lens_i, ctx, out_alphas, t);
        build_x_kernel<<<(NB * NX + 255) / 256, 256, 0, stream>>>(emb_bf, ctx, x_bf, t);
        gemm(x_bf, NX, Wih_bf, NX, b_ih, gates, NG, nullptr, gates, NG, NB, NG, NX);
        lstm_update_kernel<<<(NB * NH + 255) / 256, 256, 0, stream>>>(gates, hbuf, cbuf,
                                                                      h_bf, hnew_bf, maskf, lens_i, t);
        // scores -> directly into d_out preds[:, t, :], masked by rowscale
        gemm(hnew_bf, NH, Wword_bf, NH, b_word, nullptr, 0, maskf,
             out_preds + (size_t)t * NV, NSTEPS * NV, NB, NV, NH);
    }
}
